// GraphER_44272522887763
// MI455X (gfx1250) — compile-verified, matrix operand order checked by reading, not run on hardware
//
#include <hip/hip_runtime.h>
#include <stdint.h>

// ---------------------------------------------------------------------------
// Types for CDNA5 WMMA (wave32)
// ---------------------------------------------------------------------------
typedef __attribute__((ext_vector_type(8)))  float    v8f;
typedef __attribute__((ext_vector_type(8)))  __bf16   v8bf;
typedef __attribute__((ext_vector_type(16))) __bf16   v16bf;
typedef __attribute__((ext_vector_type(4)))  unsigned v4u;
typedef __attribute__((ext_vector_type(8)))  unsigned v8u;

#define AS_STRIDE 72   // bf16 elems; 144B row -> 16B aligned, banks 36*r mod 64 distinct
#define BS_STRIDE 72
#define ASZ (128 * AS_STRIDE)
#define BSZ (64 * BS_STRIDE)
#define USE_TDM 1      // Tensor Data Mover for the A tile

static __device__ __forceinline__ v16bf make_frag(const __bf16* p0, const __bf16* p1) {
    // Two contiguous 16B LDS reads -> ds_load_b128 x2
    v8bf a = __builtin_bit_cast(v8bf, *(const uint4*)p0);
    v8bf b = __builtin_bit_cast(v8bf, *(const uint4*)p1);
    v16bf r;
#pragma unroll
    for (int i = 0; i < 8; ++i) { r[i] = a[i]; r[8 + i] = b[i]; }
    return r;
}

// ---------------------------------------------------------------------------
// Tiled bf16 GEMM:  Dst[M,NOUT] = act( A[M,K] @ B[K,NOUT] + bias )  (bf16 out)
// Block: 256 thr = 8 waves, tile 128x64, stage BK=64 (2 WMMA k-substeps).
// Double-buffered LDS; wave0 issues TDM for stage s+1 while stage s computes;
// next B tile prefetched into registers during compute. 1 barrier per stage,
// 8 v_wmma between barriers.
// ---------------------------------------------------------------------------
template <bool RELU, int NOUT>
__global__ __launch_bounds__(256) void gemm_bf16_kernel(
    const __bf16* __restrict__ A, const __bf16* __restrict__ B,
    const float* __restrict__ bias, __bf16* __restrict__ Dst,
    int M, int K)
{
    __shared__ __bf16 As[2][ASZ];
    __shared__ __bf16 Bs[2][BSZ];

    const int tid  = threadIdx.x;
    const int lane = tid & 31;
    const int wave = tid >> 5;
    const int wr   = wave >> 1;            // 0..3 (row group)
    const int wc   = wave & 1;             // 0..1 (col group)
    const int l16  = lane & 15;
    const int koff = (lane >> 4) << 3;     // 0 or 8 (ISA 16-bit A/B layout)
    const int m0   = blockIdx.x * 128;
    const int n0   = blockIdx.y * 64;
    const int nstage = K >> 6;             // K is a multiple of 64 here

    v8f acc[2][2];
#pragma unroll
    for (int i = 0; i < 2; ++i)
#pragma unroll
        for (int j = 0; j < 2; ++j)
#pragma unroll
            for (int e = 0; e < 8; ++e) acc[i][j][e] = 0.0f;

#if !USE_TDM
    // A staging: 128 rows x 64 bf16 per stage; 4 chunks of 8 bf16 per thread
    const int arow = tid >> 3, aseg = (tid & 7) * 8;
    const __bf16* aPtr[4];
    int asOff[4];
#pragma unroll
    for (int p = 0; p < 4; ++p) {
        int r = p * 32 + arow;
        int gr = m0 + r; if (gr > M - 1) gr = M - 1;
        aPtr[p] = A + (size_t)gr * K + aseg;
        asOff[p] = r * AS_STRIDE + aseg;
    }
#endif
    // B staging: 64 rows(k) x 64 cols per stage; two 8-col chunks per thread
    const int bk = tid >> 3;               // 0..31 (chunk1 uses bk+32)
    const int bn = (tid & 7) << 3;
    const __bf16* bPtr = B + (size_t)bk * NOUT + n0 + bn;
    const int bsOff = bn * BS_STRIDE + bk; // element offset inside one buffer

    // Fragment LDS base offsets (element offsets inside a buffer)
    int apOff[2], bpOff[2];
#pragma unroll
    for (int ar = 0; ar < 2; ++ar) apOff[ar] = (wr * 32 + ar * 16 + l16) * AS_STRIDE + koff;
#pragma unroll
    for (int bc = 0; bc < 2; ++bc) bpOff[bc] = (wc * 32 + bc * 16 + l16) * BS_STRIDE + koff;

#if USE_TDM
    // Tensor DMA descriptor (CDNA5 D#, 2D tile 64x128 rows with LDS padding)
    const unsigned dim1 = (unsigned)(M - m0);                    // OOB rows read zero
    const unsigned long long gaBase =
        (unsigned long long)(uintptr_t)A + ((size_t)m0 * K) * 2; // tile start, bytes
    v8u g1;
    g1[0] = (1u << 16)        // data_size = 2 bytes
          | (1u << 20)        // pad_enable
          | (4u << 22)        // pad_interval: 32 DWORDs (= 64 bf16 = tile row)
          | (3u << 25);       // pad_amount: 4 DWORDs (= 8 bf16 -> stride 72)
    g1[1] = (64u & 0xFFFFu) << 16;                        // tensor_dim0 = 64 (lo16)
    g1[2] = ((dim1 & 0xFFFFu) << 16);                     // dim0 hi16=0 | tensor_dim1 lo16
    g1[3] = ((dim1 >> 16) & 0xFFFFu) | (64u << 16);       // tensor_dim1 hi16 | tile_dim0=64
    g1[4] = 128u;                                         // tile_dim1=128, tile_dim2=0
    g1[5] = (unsigned)K;                                  // tensor_dim0_stride lo32 (elems)
    g1[6] = 0u;
    g1[7] = 0u;
    const unsigned ldsA0 = (unsigned)(uintptr_t)(void*)&As[0][0];
    const unsigned ldsA1 = (unsigned)(uintptr_t)(void*)&As[1][0];

    #define TDM_ISSUE(stage, buf)                                              \
        do {                                                                   \
            unsigned long long ga_ = gaBase + (unsigned)(stage) * 128u;        \
            v4u g0_;                                                           \
            g0_[0] = 1u;                                                       \
            g0_[1] = (buf) ? ldsA1 : ldsA0;                                    \
            g0_[2] = (unsigned)ga_;                                            \
            g0_[3] = (unsigned)((ga_ >> 32) & 0x01FFFFFFu) | 0x80000000u;      \
            asm volatile("tensor_load_to_lds %0, %1" :: "s"(g0_), "s"(g1)      \
                         : "memory");                                          \
        } while (0)
#endif

    // ---- prologue: stage 0 into buffer 0 ----
#if USE_TDM
    if (wave == 0) TDM_ISSUE(0, 0);
#else
#pragma unroll
    for (int p = 0; p < 4; ++p) *(uint4*)&As[0][asOff[p]] = *(const uint4*)aPtr[p];
#endif
    {
        uint4 v0 = *(const uint4*)bPtr;
        uint4 v1 = *(const uint4*)(bPtr + (size_t)32 * NOUT);
        v8bf t0 = __builtin_bit_cast(v8bf, v0);
        v8bf t1 = __builtin_bit_cast(v8bf, v1);
#pragma unroll
        for (int i = 0; i < 8; ++i) {
            Bs[0][bsOff + i * BS_STRIDE]      = t0[i];
            Bs[0][bsOff + i * BS_STRIDE + 32] = t1[i];
        }
    }
#if USE_TDM
    if (wave == 0) __builtin_amdgcn_s_wait_tensorcnt(0);
#endif
    __syncthreads();

    // ---- main pipelined loop ----
    for (int s = 0; s < nstage; ++s) {
        const int cur = s & 1, nxt = cur ^ 1;
        const bool more = (s + 1 < nstage);

        uint4 nb0, nb1;
        if (more) {
#if USE_TDM
            if (wave == 0) TDM_ISSUE(s + 1, nxt);
#else
#pragma unroll
            for (int p = 0; p < 4; ++p) {
                uint4 v = *(const uint4*)(aPtr[p] + (size_t)(s + 1) * 64);
                *(uint4*)&As[nxt][asOff[p]] = v;
            }
#endif
            nb0 = *(const uint4*)(bPtr + (size_t)(s + 1) * 64 * NOUT);
            nb1 = *(const uint4*)(bPtr + (size_t)((s + 1) * 64 + 32) * NOUT);
        }

        // compute stage s: two k-substeps of 32, 4 wmma each
#pragma unroll
        for (int kk = 0; kk < 2; ++kk) {
            v16bf af[2], bfr[2];
#pragma unroll
            for (int ar = 0; ar < 2; ++ar) {
                const __bf16* p = &As[cur][apOff[ar] + kk * 32];
                af[ar] = make_frag(p, p + 16);
            }
#pragma unroll
            for (int bc = 0; bc < 2; ++bc) {
                const __bf16* p = &Bs[cur][bpOff[bc] + kk * 32];
                bfr[bc] = make_frag(p, p + 16);
            }
#pragma unroll
            for (int ar = 0; ar < 2; ++ar)
#pragma unroll
                for (int bc = 0; bc < 2; ++bc)
                    acc[ar][bc] = __builtin_amdgcn_wmma_f32_16x16x32_bf16(
                        false, af[ar], false, bfr[bc], (short)0, acc[ar][bc],
                        false, false);
        }

        if (more) {
            v8bf t0 = __builtin_bit_cast(v8bf, nb0);
            v8bf t1 = __builtin_bit_cast(v8bf, nb1);
#pragma unroll
            for (int i = 0; i < 8; ++i) {
                Bs[nxt][bsOff + i * BS_STRIDE]      = t0[i];
                Bs[nxt][bsOff + i * BS_STRIDE + 32] = t1[i];
            }
#if USE_TDM
            if (wave == 0) __builtin_amdgcn_s_wait_tensorcnt(0);
#endif
        }
        __syncthreads();
    }

    // ---- epilogue: bias + optional relu, bf16 store (fast path: full tile) ----
    const int rsub = (lane >> 4) << 3;   // C/D layout: lanes>=16 hold M=8..15
    const bool fullTile = (m0 + 128 <= M);
#pragma unroll
    for (int ar = 0; ar < 2; ++ar) {
#pragma unroll
        for (int bc = 0; bc < 2; ++bc) {
            const int col  = n0 + wc * 32 + bc * 16 + l16;
            const int row0 = m0 + wr * 32 + ar * 16 + rsub;
            const float bv = bias[col];
            __bf16* dp = Dst + (size_t)row0 * NOUT + col;
            if (fullTile) {
#pragma unroll
                for (int r = 0; r < 8; ++r) {
                    float v = acc[ar][bc][r] + bv;
                    if (RELU) v = v > 0.0f ? v : 0.0f;
                    dp[(size_t)r * NOUT] = (__bf16)v;
                }
            } else {
#pragma unroll
                for (int r = 0; r < 8; ++r) {
                    if (row0 + r < M) {
                        float v = acc[ar][bc][r] + bv;
                        if (RELU) v = v > 0.0f ? v : 0.0f;
                        dp[(size_t)r * NOUT] = (__bf16)v;
                    }
                }
            }
        }
    }
}

// ---------------------------------------------------------------------------
// Small helper kernels
// ---------------------------------------------------------------------------
__global__ void cvt_f32_bf16_kernel(const float* __restrict__ s, __bf16* __restrict__ d,
                                    long n) {
    long i = (long)blockIdx.x * blockDim.x + threadIdx.x;
    if (i < n) d[i] = (__bf16)s[i];
}

// h = x + agg  (bf16 <- bf16 + f32)
__global__ void hprep_kernel(const __bf16* __restrict__ x, const float* __restrict__ agg,
                             __bf16* __restrict__ h, long n) {
    long i = (long)blockIdx.x * blockDim.x + threadIdx.x;
    if (i < n) h[i] = (__bf16)((float)x[i] + agg[i]);
}

// agg[dst] += x[src]  -- 8 features per thread, f32 atomics (accum fits in L2)
__global__ void scatter_kernel(const __bf16* __restrict__ x, const int* __restrict__ ei,
                               float* __restrict__ agg, int E, int Dl, int lchunks) {
    long idx = (long)blockIdx.x * blockDim.x + threadIdx.x;
    long total = (long)E << lchunks;
    if (idx >= total) return;
    int e  = (int)(idx >> lchunks);
    int ch = (int)(idx & ((1 << lchunks) - 1));
    int s = ei[e];
    int d = ei[E + e];
    uint4 v = *(const uint4*)(x + (size_t)s * Dl + ch * 8);
    v8bf t8 = __builtin_bit_cast(v8bf, v);
    float* dst = agg + (size_t)d * Dl + ch * 8;
#pragma unroll
    for (int i = 0; i < 8; ++i)
        __hip_atomic_fetch_add(&dst[i], (float)t8[i], __ATOMIC_RELAXED,
                               __HIP_MEMORY_SCOPE_AGENT);
}

// time MLP (1 -> 512 -> 512), f32
__global__ void time_hidden_kernel(const float* __restrict__ t, const float* __restrict__ w1,
                                   const float* __restrict__ b1, float* __restrict__ th) {
    int j = threadIdx.x;  // 512 threads, 1 block
    float v = t[0] * w1[j] + b1[j];
    th[j] = v > 0.0f ? v : 0.0f;
}
__global__ void time_out_kernel(const float* __restrict__ th, const float* __restrict__ w2,
                                const float* __restrict__ b2, float* __restrict__ temb) {
    int j = blockIdx.x * blockDim.x + threadIdx.x;  // 512
    float s = b2[j];
    for (int i = 0; i < 512; ++i) s += th[i] * w2[(size_t)i * 512 + j];
    temb[j] = s;
}

// first-edge representation: ff[0:512]=xu+xv, ff[512:1024]=|xu-xv|
__global__ void first_feat_kernel(const __bf16* __restrict__ x, const int* __restrict__ fe,
                                  float* __restrict__ ff) {
    int d = threadIdx.x;  // 512 threads, 1 block
    int u = fe[0], v = fe[1];
    float xu = (float)x[(size_t)u * 512 + d];
    float xv = (float)x[(size_t)v * 512 + d];
    ff[d]       = xu + xv;
    ff[512 + d] = fabsf(xu - xv);
}

// partial[j] = b1[j] + ff . W1[0:1024, j] + temb . W1[2048:2560, j]
// (exact hoist of the row-invariant part of the head GEMM; K shrinks 2560->1024)
__global__ void partial_kernel(const float* __restrict__ ff, const float* __restrict__ temb,
                               const float* __restrict__ w1, const float* __restrict__ b1,
                               float* __restrict__ partial) {
    int j = blockIdx.x * blockDim.x + threadIdx.x;  // 512
    float s = b1[j];
    for (int i = 0; i < 1024; ++i) s += ff[i] * w1[(size_t)i * 512 + j];
    for (int i = 0; i < 512; ++i)  s += temb[i] * w1[(size_t)(2048 + i) * 512 + j];
    partial[j] = s;
}

// candidate edge features -> featA[c, 0:1024] (bf16)
__global__ void cand_feat_kernel(const __bf16* __restrict__ x, const int* __restrict__ ce,
                                 __bf16* __restrict__ featA, int C) {
    long idx = (long)blockIdx.x * blockDim.x + threadIdx.x;
    if (idx >= (long)C * 512) return;
    int c = (int)(idx >> 9);
    int d = (int)(idx & 511);
    int u = ce[2 * c], v = ce[2 * c + 1];
    float xu = (float)x[(size_t)u * 512 + d];
    float xv = (float)x[(size_t)v * 512 + d];
    featA[(size_t)c * 1024 + d]       = (__bf16)(xu + xv);
    featA[(size_t)c * 1024 + 512 + d] = (__bf16)fabsf(xu - xv);
}

// logits[c] = hid[c,:] . w2 + b2   (one wave per row, wave32 shuffle reduce)
__global__ void logits_kernel(const __bf16* __restrict__ hid, const float* __restrict__ w2,
                              const float* __restrict__ b2, float* __restrict__ out, int C) {
    int gw   = (int)((blockIdx.x * (long)blockDim.x + threadIdx.x) >> 5);
    int lane = threadIdx.x & 31;
    if (gw >= C) return;
    float s = 0.0f;
    for (int j = lane; j < 512; j += 32) s += (float)hid[(size_t)gw * 512 + j] * w2[j];
#pragma unroll
    for (int off = 16; off > 0; off >>= 1) s += __shfl_down(s, off, 32);
    if (lane == 0) out[gw] = s + b2[0];
}

// ---------------------------------------------------------------------------
// Launcher
// ---------------------------------------------------------------------------
extern "C" void kernel_launch(void* const* d_in, const int* in_sizes, int n_in,
                              void* d_out, int out_size, void* d_ws, size_t ws_size,
                              hipStream_t stream) {
    (void)in_sizes; (void)n_in; (void)out_size; (void)ws_size;
    const int N = 50000, E = 1600000, C = 50000, D = 128, H = 512;

    const float* x_in  = (const float*)d_in[0];
    const int*   ei    = (const int*)d_in[1];
    const int*   fe    = (const int*)d_in[2];
    const int*   ce    = (const int*)d_in[3];
    const float* t_in  = (const float*)d_in[4];
    const float* gw1[3] = {(const float*)d_in[5],  (const float*)d_in[9],  (const float*)d_in[13]};
    const float* gb1[3] = {(const float*)d_in[6],  (const float*)d_in[10], (const float*)d_in[14]};
    const float* gw2[3] = {(const float*)d_in[7],  (const float*)d_in[11], (const float*)d_in[15]};
    const float* gb2[3] = {(const float*)d_in[8],  (const float*)d_in[12], (const float*)d_in[16]};
    const float* pred_w1 = (const float*)d_in[17];
    const float* pred_b1 = (const float*)d_in[18];
    const float* pred_w2 = (const float*)d_in[19];
    const float* pred_b2 = (const float*)d_in[20];
    const float* time_w1 = (const float*)d_in[21];
    const float* time_b1 = (const float*)d_in[22];
    const float* time_w2 = (const float*)d_in[23];
    const float* time_b2 = (const float*)d_in[24];
    float* logits = (float*)d_out;

    // ---- carve workspace (256B aligned) ----
    uint8_t* p = (uint8_t*)d_ws;
    auto carve = [&](size_t bytes) -> void* {
        void* r = (void*)p;
        p += (bytes + 255) & ~(size_t)255;
        return r;
    };
    __bf16* xb    = (__bf16*)carve((size_t)N * 512 * 2);   // node features (bf16)
    __bf16* hb    = (__bf16*)carve((size_t)N * 512 * 2);   // x + agg
    __bf16* tb    = (__bf16*)carve((size_t)N * 512 * 2);   // mid activation
    float*  agg   = (float*) carve((size_t)N * 512 * 4);   // scatter accumulator
    __bf16* featA = (__bf16*)carve((size_t)C * 1024 * 2);  // candidate features
    __bf16* hid   = (__bf16*)carve((size_t)C * 512 * 2);   // head hidden
    __bf16* w1b[3], *w2b[3];
    w1b[0] = (__bf16*)carve((size_t)D * H * 2);
    w2b[0] = (__bf16*)carve((size_t)H * H * 2);
    for (int l = 1; l < 3; ++l) {
        w1b[l] = (__bf16*)carve((size_t)H * H * 2);
        w2b[l] = (__bf16*)carve((size_t)H * H * 2);
    }
    __bf16* pW1b    = (__bf16*)carve((size_t)1024 * 512 * 2);  // pred_w1 rows 1024..2047
    float*  ff      = (float*)carve(1024 * 4);
    float*  th      = (float*)carve(512 * 4);
    float*  temb    = (float*)carve(512 * 4);
    float*  partial = (float*)carve(512 * 4);

    const dim3 blk(256);
    auto blocks1d = [](long n) { return dim3((unsigned)((n + 255) / 256)); };

    // ---- convert inputs / weights to bf16 ----
    hipLaunchKernelGGL(cvt_f32_bf16_kernel, blocks1d((long)N * D), blk, 0, stream, x_in, xb, (long)N * D);
    hipLaunchKernelGGL(cvt_f32_bf16_kernel, blocks1d((long)D * H), blk, 0, stream, gw1[0], w1b[0], (long)D * H);
    hipLaunchKernelGGL(cvt_f32_bf16_kernel, blocks1d((long)H * H), blk, 0, stream, gw2[0], w2b[0], (long)H * H);
    for (int l = 1; l < 3; ++l) {
        hipLaunchKernelGGL(cvt_f32_bf16_kernel, blocks1d((long)H * H), blk, 0, stream, gw1[l], w1b[l], (long)H * H);
        hipLaunchKernelGGL(cvt_f32_bf16_kernel, blocks1d((long)H * H), blk, 0, stream, gw2[l], w2b[l], (long)H * H);
    }
    hipLaunchKernelGGL(cvt_f32_bf16_kernel, blocks1d((long)1024 * 512), blk, 0, stream,
                       pred_w1 + (size_t)1024 * 512, pW1b, (long)1024 * 512);

    // ---- time MLP (tiny, f32) ----
    hipLaunchKernelGGL(time_hidden_kernel, dim3(1), dim3(512), 0, stream, t_in, time_w1, time_b1, th);
    hipLaunchKernelGGL(time_out_kernel, dim3(2), dim3(256), 0, stream, th, time_w2, time_b2, temb);

    // ---- 3 GIN layers ----
    for (int l = 0; l < 3; ++l) {
        const int Din = (l == 0) ? D : H;
        const int lch = (Din == 128) ? 4 : 6;  // log2(Din/8)
        hipMemsetAsync(agg, 0, (size_t)N * Din * 4, stream);
        hipLaunchKernelGGL(scatter_kernel, blocks1d((long)E * (Din / 8)), blk, 0, stream,
                           xb, ei, agg, E, Din, lch);
        hipLaunchKernelGGL(hprep_kernel, blocks1d((long)N * Din), blk, 0, stream,
                           xb, agg, hb, (long)N * Din);
        dim3 g1((N + 127) / 128, H / 64);
        hipLaunchKernelGGL((gemm_bf16_kernel<true, 512>),  g1, blk, 0, stream, hb, w1b[l], gb1[l], tb, N, Din);
        hipLaunchKernelGGL((gemm_bf16_kernel<false, 512>), g1, blk, 0, stream, tb, w2b[l], gb2[l], xb, N, H);
    }

    // ---- head: hoisted row-invariant part + K=1024 WMMA GEMM ----
    hipLaunchKernelGGL(first_feat_kernel, dim3(1), dim3(512), 0, stream, xb, fe, ff);
    hipLaunchKernelGGL(partial_kernel, dim3(2), dim3(256), 0, stream, ff, temb, pred_w1, pred_b1, partial);
    hipLaunchKernelGGL(cand_feat_kernel, blocks1d((long)C * 512), blk, 0, stream, xb, ce, featA, C);
    dim3 gh((C + 127) / 128, H / 64);
    hipLaunchKernelGGL((gemm_bf16_kernel<true, 512>), gh, blk, 0, stream, featA, pW1b, partial, hid, C, 1024);
    hipLaunchKernelGGL(logits_kernel, blocks1d((long)C * 32), blk, 0, stream, hid, pred_w2, pred_b2, logits, C);
}
